// VectorQuantizer_84731114815637
// MI455X (gfx1250) — matrix operand order repaired and checked by reading, not compiled, not run on hardware
//
#include <hip/hip_runtime.h>

typedef __bf16 bf16_t;
typedef __attribute__((ext_vector_type(16))) __bf16 v16bf;
typedef __attribute__((ext_vector_type(8)))  __bf16 v8bf;
typedef __attribute__((ext_vector_type(4)))  __bf16 v4bf;
typedef __attribute__((ext_vector_type(8)))  float  v8f;
typedef __attribute__((ext_vector_type(4)))  float  v4f;
typedef __attribute__((ext_vector_type(4)))  unsigned int v4u;
typedef __attribute__((ext_vector_type(8)))  int    v8i;
typedef __attribute__((ext_vector_type(4)))  int    v4i;

#define KCODES 1024
#define DIM    128
#define MT     128          // rows per block (8 waves x 16 rows)
#define KC     64           // codes per LDS chunk
#define NCHUNK (KCODES / KC)
#define LDB    (DIM + 8)    // bf16 row stride (+16B) -> rows hit distinct LDS banks
#define NROWS  (16 * 4096)
#define NELEM  ((size_t)NROWS * (size_t)DIM)

// ---------------------------------------------------------------------------
// TDM: DMA one [KC x DIM] bf16 code chunk from global into LDS, inserting
// 16B of padding after every 256B row so the LDS image has stride LDB.
//   pad_interval code 5 -> pad after 64 DWORDs (= 128 bf16 = one row)
//   pad_amount   code 3 -> insert 4 DWORDs (= 8 bf16 = LDB - DIM)
__device__ __forceinline__ void tdm_load_codes(const bf16_t* gsrc, unsigned lds_off) {
  unsigned long long ga = (unsigned long long)(uintptr_t)gsrc;
  v4u g0 = { 1u,                                   // count=1 (valid user D#)
             lds_off,                              // lds_addr
             (unsigned)(ga & 0xFFFFFFFFu),         // global_addr[31:0]
             (unsigned)((ga >> 32) & 0x01FFFFFFu)  // global_addr[56:32]
               | (2u << 30) };                     // type=2 ("image")
  v8i g1 = { (int)((1u << 16)      // data_size = 2 bytes
                 | (1u << 20)      // pad_enable
                 | (5u << 22)      // pad_interval: 64 DWORDs
                 | (3u << 25)),    // pad_amount : 4 DWORDs
             (int)((unsigned)DIM << 16),   // tensor_dim0[15:0] @ bits 63:48
             (int)((unsigned)KC  << 16),   // tensor_dim0 hi=0, tensor_dim1=KC
             (int)((unsigned)DIM << 16),   // tensor_dim1 hi=0, tile_dim0=DIM
             (int)KC,                      // tile_dim1=KC, tile_dim2=0
             (int)DIM,                     // tensor_dim0_stride = DIM
             0, 0 };
  v4i z4 = { 0, 0, 0, 0 };
  v8i z8 = { 0, 0, 0, 0, 0, 0, 0, 0 };
  __builtin_amdgcn_tensor_load_to_lds(g0, g1, z4, z4, z8, 0);
}

// B fragment (32x16 bf16): N = lane&15; lanes 0-15 hold K {0..7,16..23}, 16-31 +8.
__device__ __forceinline__ v16bf frag_b(const bf16_t* rowp, int db, int half) {
  v8bf lo = *(const v8bf*)(rowp + db + half * 8);
  v8bf hi = *(const v8bf*)(rowp + db + half * 8 + 16);
  return __builtin_shufflevector(lo, hi, 0,1,2,3,4,5,6,7,8,9,10,11,12,13,14,15);
}

__device__ __forceinline__ v8bf cvt8(v4f a, v4f b) {
  v8bf r = { (bf16_t)a.x, (bf16_t)a.y, (bf16_t)a.z, (bf16_t)a.w,
             (bf16_t)b.x, (bf16_t)b.y, (bf16_t)b.z, (bf16_t)b.w };
  return r;
}

// ---------------------------------------------------------------------------
// One-shot prep: fp32 codebooks -> bf16 images + fp32 squared-norm tables.
__global__ __launch_bounds__(256) void vq_prep_kernel(
    const float* __restrict__ emb1, const float* __restrict__ emb2,
    bf16_t* __restrict__ bf1, bf16_t* __restrict__ bf2,
    float* __restrict__ esq1, float* __restrict__ esq2) {
  int t    = threadIdx.x;
  int rowg = blockIdx.x * 64 + (t >> 2);      // 4 threads per code row
  int quarter = t & 3;
  const float* src; bf16_t* dst; float* esq;
  if (rowg < KCODES) {
    src = emb1 + (size_t)rowg * DIM; dst = bf1 + (size_t)rowg * DIM; esq = esq1 + rowg;
  } else {
    int r = rowg - KCODES;
    src = emb2 + (size_t)r * DIM;    dst = bf2 + (size_t)r * DIM;    esq = esq2 + r;
  }
  float ssq = 0.f;
#pragma unroll
  for (int j = 0; j < 8; ++j) {
    int c = quarter * 32 + j * 4;
    v4f v = *(const v4f*)(src + c);
    ssq += v.x * v.x + v.y * v.y + v.z * v.z + v.w * v.w;
    v4bf bv = { (bf16_t)v.x, (bf16_t)v.y, (bf16_t)v.z, (bf16_t)v.w };
    *(v4bf*)(dst + c) = bv;
  }
  ssq += __shfl_xor(ssq, 1);
  ssq += __shfl_xor(ssq, 2);
  if (quarter == 0) *esq = ssq;
}

// ---------------------------------------------------------------------------
template <bool STAGE2>
__global__ __launch_bounds__(256) void vq_stage_kernel(
    const float*  __restrict__ latents,
    const bf16_t* __restrict__ embbf,     // bf16 codebook (this stage)
    const float*  __restrict__ esq,       // ||e||^2 table (this stage)
    const float*  __restrict__ emb,       // fp32 codebook (this stage, writeout)
    const float*  __restrict__ emb_prev,  // fp32 emb1 (stage 2)
    int*          __restrict__ idx_io,    // stage1: write idx; stage2: read idx1
    float*        __restrict__ qout,      // stage2: quantized out
    float*        __restrict__ partials)  // stage2: per-block loss partials
{
  __shared__ alignas(16) bf16_t sB[2][KC][LDB];
  __shared__ float sRed[8];

  const int tid     = threadIdx.x;
  const int lane    = tid & 31;
  const int wave    = tid >> 5;
  const int half    = lane >> 4;
  const int mrow    = lane & 15;
  const int rowbase = blockIdx.x * MT;
  const int myrow   = rowbase + wave * 16 + mrow;

  // ---- build this lane's A fragments straight from global (no LDS pass) ---
  // A (16x32 bf16): lanes 0-15 (M=lane) hold K {0..7,16..23}; lanes 16-31 +8.
  v16bf afr[4];
  {
    const float* xp = latents + (size_t)myrow * DIM;
    const float* qp = nullptr;
    if (STAGE2) qp = emb_prev + (size_t)idx_io[myrow] * DIM;
#pragma unroll
    for (int kk = 0; kk < 4; ++kk) {
      int d0 = kk * 32 + half * 8;
      int d1 = d0 + 16;
      v4f a0 = *(const v4f*)(xp + d0), a1 = *(const v4f*)(xp + d0 + 4);
      v4f b0 = *(const v4f*)(xp + d1), b1 = *(const v4f*)(xp + d1 + 4);
      if (STAGE2) {
        a0 -= *(const v4f*)(qp + d0);  a1 -= *(const v4f*)(qp + d0 + 4);
        b0 -= *(const v4f*)(qp + d1);  b1 -= *(const v4f*)(qp + d1 + 4);
      }
      afr[kk] = __builtin_shufflevector(cvt8(a0, a1), cvt8(b0, b1),
                                        0,1,2,3,4,5,6,7,8,9,10,11,12,13,14,15);
    }
  }

  float minv[8];
  int   mini[8];
#pragma unroll
  for (int r = 0; r < 8; ++r) { minv[r] = 3.4e38f; mini[r] = 0; }

  const unsigned ldsB0 = (unsigned)(uintptr_t)&sB[0][0][0];
  const unsigned ldsB1 = (unsigned)(uintptr_t)&sB[1][0][0];

  // prime the pipeline: TDM chunk 0 into buffer 0 (one DMA per block, wave 0)
  if (wave == 0) tdm_load_codes(embbf, ldsB0);

  for (int c = 0; c < NCHUNK; ++c) {
    if (wave == 0) {
      if (c + 1 < NCHUNK) {
        tdm_load_codes(embbf + (size_t)(c + 1) * KC * DIM, (c & 1) ? ldsB0 : ldsB1);
        __builtin_amdgcn_s_wait_tensorcnt(1);  // in-order: chunk c is resident
      } else {
        __builtin_amdgcn_s_wait_tensorcnt(0);
      }
    }
    __syncthreads();                            // chunk c visible to all waves

    const bf16_t* bbase = &sB[c & 1][0][0];
#pragma unroll
    for (int ct = 0; ct < 4; ++ct) {
      int col = ct * 16 + mrow;
      const bf16_t* bp = bbase + col * LDB;
      v8f acc = {};
#pragma unroll
      for (int kk = 0; kk < 4; ++kk) {
        v16bf b = frag_b(bp, kk * 32, half);
        acc = __builtin_amdgcn_wmma_f32_16x16x32_bf16(
            false, afr[kk], false, b, (short)0, acc, false, false);
      }
      int   gid = c * KC + col;
      float es  = esq[gid];
#pragma unroll
      for (int r = 0; r < 8; ++r) {
        float s = es - 2.0f * acc[r];          // ||e||^2 - 2 x.e
        if (s < minv[r]) { minv[r] = s; mini[r] = gid; }
      }
    }
    __syncthreads();   // done reading buffer (c&1) before TDM refills it
  }

  // ---- argmin butterfly within each 16-lane half (rows split by half) -----
#pragma unroll
  for (int m = 1; m <= 8; m <<= 1) {
#pragma unroll
    for (int r = 0; r < 8; ++r) {
      float ov = __shfl_xor(minv[r], m);
      int   oi = __shfl_xor(mini[r], m);
      if (ov < minv[r] || (ov == minv[r] && oi < mini[r])) {
        minv[r] = ov; mini[r] = oi;
      }
    }
  }

  if (!STAGE2) {
    if (lane == 0 || lane == 16) {
      int rb = rowbase + wave * 16 + half * 8;
#pragma unroll
      for (int r = 0; r < 8; ++r) idx_io[rb + r] = mini[r];
    }
  } else {
    // ---- fused writeout: quantized = emb1[i1] + emb2[i2]; loss partials ---
    float lacc = 0.f;
    int   c0   = lane * 4;   // 32 lanes x float4 = 128 cols
#pragma unroll
    for (int r = 0; r < 8; ++r) {
      int iLo = __shfl(mini[r], 0);
      int iHi = __shfl(mini[r], 16);
      int rl  = rowbase + wave * 16 + r;
      int rh  = rl + 8;
      {
        v4f e2 = *(const v4f*)(emb      + (size_t)iLo * DIM + c0);
        v4f q1 = *(const v4f*)(emb_prev + (size_t)idx_io[rl] * DIM + c0);
        v4f la = *(const v4f*)(latents  + (size_t)rl * DIM + c0);
        v4f q  = e2 + q1;
        *(v4f*)(qout + (size_t)rl * DIM + c0) = q;
        v4f d = q - la;
        lacc += d.x * d.x + d.y * d.y + d.z * d.z + d.w * d.w;
      }
      {
        v4f e2 = *(const v4f*)(emb      + (size_t)iHi * DIM + c0);
        v4f q1 = *(const v4f*)(emb_prev + (size_t)idx_io[rh] * DIM + c0);
        v4f la = *(const v4f*)(latents  + (size_t)rh * DIM + c0);
        v4f q  = e2 + q1;
        *(v4f*)(qout + (size_t)rh * DIM + c0) = q;
        v4f d = q - la;
        lacc += d.x * d.x + d.y * d.y + d.z * d.z + d.w * d.w;
      }
    }
#pragma unroll
    for (int m = 16; m >= 1; m >>= 1) lacc += __shfl_xor(lacc, m);
    if (lane == 0) sRed[wave] = lacc;
    __syncthreads();
    if (tid == 0) {
      float s = 0.f;
#pragma unroll
      for (int w = 0; w < 8; ++w) s += sRed[w];
      partials[blockIdx.x] = s;
    }
  }
}

__global__ __launch_bounds__(256) void vq_finalize_kernel(
    const float* __restrict__ partials, float* __restrict__ loss_out) {
  __shared__ float red[256];
  int t = threadIdx.x;
  red[t] = partials[t] + partials[t + 256];   // 512 partials, fixed order
  __syncthreads();
  for (int s = 128; s > 0; s >>= 1) {
    if (t < s) red[t] += red[t + s];
    __syncthreads();
  }
  if (t == 0) {
    // vq_loss = beta*mse + mse = 1.25 * sum_sq / (N*D)
    *loss_out = red[0] * 1.25f / (float)NELEM;
  }
}

// ---------------------------------------------------------------------------
extern "C" void kernel_launch(void* const* d_in, const int* in_sizes, int n_in,
                              void* d_out, int out_size, void* d_ws, size_t ws_size,
                              hipStream_t stream) {
  const float* latents = (const float*)d_in[0];
  const float* emb1    = (const float*)d_in[1];
  const float* emb2    = (const float*)d_in[2];
  float* out = (float*)d_out;

  char* ws = (char*)d_ws;
  int*    idx1     = (int*)ws;                                   ws += (size_t)NROWS * 4;
  float*  partials = (float*)ws;                                 ws += 512 * 4;
  bf16_t* bf1      = (bf16_t*)ws;                                ws += (size_t)KCODES * DIM * 2;
  bf16_t* bf2      = (bf16_t*)ws;                                ws += (size_t)KCODES * DIM * 2;
  float*  esq1     = (float*)ws;                                 ws += KCODES * 4;
  float*  esq2     = (float*)ws;

  const int blocks = NROWS / MT;  // 512
  vq_prep_kernel<<<2 * KCODES / 64, 256, 0, stream>>>(emb1, emb2, bf1, bf2, esq1, esq2);
  vq_stage_kernel<false><<<blocks, 256, 0, stream>>>(
      latents, bf1, esq1, emb1, nullptr, idx1, nullptr, nullptr);
  vq_stage_kernel<true><<<blocks, 256, 0, stream>>>(
      latents, bf2, esq2, emb2, emb1, idx1, out, partials);
  vq_finalize_kernel<<<1, 256, 0, stream>>>(partials, out + NELEM);
}
